// anomaly_anticipation_80049600463197
// MI455X (gfx1250) — compile-verified, matrix operand order checked by reading, not run on hardware
//
#include <hip/hip_runtime.h>
#include <stdint.h>

// ---------------------------------------------------------------------------
// GCN forward for MI455X (gfx1250, wave32).  GEMMs run on v_wmma_f32_16x16x32_bf16
// (bf16 A/B, f32 accumulate); graph aggregation uses float4 gathers + f32
// global atomics (h tiles fit in the 192MB L2); tiny MLP head is fused.
// ---------------------------------------------------------------------------

typedef __attribute__((ext_vector_type(16))) __bf16 v16bf;
typedef __attribute__((ext_vector_type(2)))  __bf16 v2bf;
typedef __attribute__((ext_vector_type(8)))  float  v8f;

union FragBF { v16bf bf; unsigned int u[8]; };
union AccF   { v8f v; float f[8]; };

__device__ __forceinline__ unsigned int pack2_bf16(float a, float b) {
  // native fptrunc f32 -> bf16 (RNE); bf16 is a legal type on gfx1250
  v2bf r = { (__bf16)a, (__bf16)b };
  return __builtin_bit_cast(unsigned int, r);
}

__device__ __forceinline__ int imin(int a, int b) { return a < b ? a : b; }

// ---------------------------------------------------------------------------
// C[M,F] = A[M,K] @ W[K,F]   (fp32 in/out, bf16 WMMA inside)
// Block: 128 threads = 4 waves. Tile: 64 rows x 32 cols, K step 32.
// Wave w owns rows [w*16, w*16+16) and two 16x16 accumulators (col halves).
// Interior tiles take an unguarded float4 staging path (block-uniform branch);
// edge tiles use branchless clamped loads + select (no exec-mask churn).
// ---------------------------------------------------------------------------
#define TM 64
#define TN 32
#define TK 32

__global__ __launch_bounds__(128)
void gemm_wmma_bf16(const float* __restrict__ A, const float* __restrict__ W,
                    float* __restrict__ C, int M, int K, int F) {
  // LDS tiles: each uint packs two consecutive-K bf16 values.
  __shared__ unsigned int As[TM][TK / 2];  // [row][k/2]
  __shared__ unsigned int Bs[TN][TK / 2];  // [col][k/2]  (W transposed)

  const int t    = threadIdx.x;      // 0..127
  const int lane = t & 31;
  const int wave = t >> 5;           // 0..3
  const int hlf  = lane >> 4;        // 0/1 : which K-half this lane holds
  const int l16  = lane & 15;
  const int rowBase = blockIdx.x * TM;
  const int colBase = blockIdx.y * TN;
  const bool fullM = (rowBase + TM) <= M;
  const bool fullN = (colBase + TN) <= F;

  v8f acc0 = {};
  v8f acc1 = {};

  const int nk = (K + TK - 1) / TK;
  for (int kt = 0; kt < nk; ++kt) {
    const int k0 = kt * TK;
    const bool fullK = (k0 + TK) <= K;

    // ---- stage A tile: 64 rows x 16 packed uints --------------------------
    if (fullM && fullK) {
      // interior: 512 float4 loads, 4 per thread, coalesced, no guards
      #pragma unroll
      for (int i = 0; i < 4; ++i) {
        int linear = i * 128 + t;            // 0..511
        int r = linear >> 3;                 // row 0..63
        int q = linear & 7;                  // float4 index within 32-K span
        const float* ap = A + (long)(rowBase + r) * K + k0 + q * 4;
        const float4 v = *(const float4*)ap;
        // prefetch next K tile of A (global_prefetch_b8) while WMMA runs
        if (kt + 1 < nk) __builtin_prefetch(ap + TK, 0, 0);
        As[r][q * 2]     = pack2_bf16(v.x, v.y);
        As[r][q * 2 + 1] = pack2_bf16(v.z, v.w);
      }
    } else {
      // edge tile: branchless — clamp address into bounds, select 0 after
      #pragma unroll
      for (int i = 0; i < 8; ++i) {
        int linear = i * 128 + t;            // 0..1023
        int r  = linear >> 4;
        int k2 = linear & 15;
        int gm = rowBase + r;
        int gk = k0 + 2 * k2;
        int cm = imin(gm, M - 1);
        int c0 = imin(gk, K - 1);
        int c1 = imin(gk + 1, K - 1);
        float v0 = A[(long)cm * K + c0];
        float v1 = A[(long)cm * K + c1];
        float a0 = (gm < M && gk < K)     ? v0 : 0.0f;
        float a1 = (gm < M && gk + 1 < K) ? v1 : 0.0f;
        As[r][k2] = pack2_bf16(a0, a1);
      }
    }

    // ---- stage W^T tile: 32 cols x 16 packed uints ------------------------
    if (fullN && fullK) {
      #pragma unroll
      for (int i = 0; i < 4; ++i) {
        int linear = i * 128 + t;            // 0..511
        int n  = linear & 31;                // col (fastest -> coalesced)
        int k2 = linear >> 5;                // 0..15
        const float* wp = W + (long)(k0 + 2 * k2) * F + colBase + n;
        Bs[n][k2] = pack2_bf16(wp[0], wp[F]);
      }
    } else {
      #pragma unroll
      for (int i = 0; i < 4; ++i) {
        int linear = i * 128 + t;
        int n  = linear & 31;
        int k2 = linear >> 5;
        int gn = colBase + n;
        int gk = k0 + 2 * k2;
        int cn = imin(gn, F - 1);
        int c0 = imin(gk, K - 1);
        int c1 = imin(gk + 1, K - 1);
        float v0 = W[(long)c0 * F + cn];
        float v1 = W[(long)c1 * F + cn];
        float b0 = (gn < F && gk < K)     ? v0 : 0.0f;
        float b1 = (gn < F && gk + 1 < K) ? v1 : 0.0f;
        Bs[n][k2] = pack2_bf16(b0, b1);
      }
    }
    __syncthreads();

    // ---- build fragments per the 16-bit A/B VGPR layout -------------------
    // VGPR v holds K pair at  (v/4)*16 + half*8 + (v%4)*2  -> /2 index below.
    FragBF a, bA, bB;
    const int m = wave * 16 + l16;
    #pragma unroll
    for (int v = 0; v < 8; ++v) {
      int idx = (v >> 2) * 8 + hlf * 4 + (v & 3);
      a.u[v]  = As[m][idx];
      bA.u[v] = Bs[l16][idx];
      bB.u[v] = Bs[16 + l16][idx];
    }

    acc0 = __builtin_amdgcn_wmma_f32_16x16x32_bf16(false, a.bf, false, bA.bf,
                                                   (short)0, acc0, false, false);
    acc1 = __builtin_amdgcn_wmma_f32_16x16x32_bf16(false, a.bf, false, bB.bf,
                                                   (short)0, acc1, false, false);
    __syncthreads();
  }

  // ---- store: VGPR r -> row (half*8 + r), col = lane%16 -------------------
  AccF c0, c1;
  c0.v = acc0;
  c1.v = acc1;
  #pragma unroll
  for (int r = 0; r < 8; ++r) {
    int gm = rowBase + wave * 16 + hlf * 8 + r;
    if (gm < M) {
      int gn0 = colBase + l16;
      int gn1 = colBase + 16 + l16;
      if (gn0 < F) C[(long)gm * F + gn0] = c0.f[r];
      if (gn1 < F) C[(long)gm * F + gn1] = c1.f[r];
    }
  }
}

// ---------------------------------------------------------------------------
// small utility kernels
// ---------------------------------------------------------------------------
__global__ void fill_f32(float* __restrict__ p, long n, float val) {
  long i = blockIdx.x * (long)blockDim.x + threadIdx.x;
  long stride = (long)gridDim.x * blockDim.x;
  for (; i < n; i += stride) p[i] = val;
}

__global__ void deg_accum(const int* __restrict__ row, float* __restrict__ deg, int E) {
  int i = blockIdx.x * blockDim.x + threadIdx.x;
  int stride = gridDim.x * blockDim.x;
  for (; i < E; i += stride) atomicAdd(&deg[row[i]], 1.0f);
}

__global__ void make_dis(float* __restrict__ d, int n) {
  int i = blockIdx.x * blockDim.x + threadIdx.x;
  int stride = gridDim.x * blockDim.x;
  for (; i < n; i += stride) {
    float v = d[i];
    d[i] = v > 0.0f ? rsqrtf(v) : 0.0f;
  }
}

// edge gather + normalized scatter-add; 4 features per work item (F % 4 == 0)
__global__ void scatter_edges(const float* __restrict__ h, const int* __restrict__ row,
                              const int* __restrict__ col, const float* __restrict__ dis,
                              float* __restrict__ agg, int E, int F) {
  const int chunks = F >> 2;
  const long total = (long)E * chunks;
  long i = blockIdx.x * (long)blockDim.x + threadIdx.x;
  long stride = (long)gridDim.x * blockDim.x;
  for (; i < total; i += stride) {
    int e = (int)(i / chunks);
    int c = (int)(i % chunks);
    int r  = row[e];
    int cl = col[e];
    float nrm = dis[r] * dis[cl];
    const float4 v = *(const float4*)(h + (long)cl * F + 4 * c);
    float* dst = agg + (long)r * F + 4 * c;
    atomicAdd(dst + 0, v.x * nrm);
    atomicAdd(dst + 1, v.y * nrm);
    atomicAdd(dst + 2, v.z * nrm);
    atomicAdd(dst + 3, v.w * nrm);
  }
}

// out = relu?( agg + h * dis^2 (self-loop) + bias )
__global__ void finalize_layer(const float* __restrict__ agg, const float* __restrict__ h,
                               const float* __restrict__ dis, const float* __restrict__ b,
                               float* __restrict__ out, int n, int F, int do_relu) {
  long total = (long)n * F;
  long i = blockIdx.x * (long)blockDim.x + threadIdx.x;
  long stride = (long)gridDim.x * blockDim.x;
  for (; i < total; i += stride) {
    int node = (int)(i / F);
    int f = (int)(i % F);
    float d = dis[node];
    float v = agg[i] + h[i] * d * d + b[f];
    if (do_relu) v = fmaxf(v, 0.0f);
    out[i] = v;
  }
}

// fused 32->16->1 head (weights staged in LDS), one thread per node
__global__ void head_mlp(const float* __restrict__ h, const float* __restrict__ Wf1,
                         const float* __restrict__ bf1, const float* __restrict__ Wf2,
                         const float* __restrict__ bf2, float* __restrict__ out, int n) {
  __shared__ float w1[32 * 16];
  __shared__ float b1s[16];
  __shared__ float w2s[16];
  __shared__ float b2s;
  int t = threadIdx.x;
  for (int i = t; i < 32 * 16; i += blockDim.x) w1[i] = Wf1[i];
  if (t < 16) { b1s[t] = bf1[t]; w2s[t] = Wf2[t]; }
  if (t == 0) b2s = bf2[0];
  __syncthreads();
  int i = blockIdx.x * blockDim.x + t;
  if (i >= n) return;
  float hin[32];
  #pragma unroll
  for (int k = 0; k < 32; ++k) hin[k] = h[(long)i * 32 + k];
  float acc = b2s;
  #pragma unroll
  for (int j = 0; j < 16; ++j) {
    float s = b1s[j];
    #pragma unroll
    for (int k = 0; k < 32; ++k) s += hin[k] * w1[k * 16 + j];
    acc += s * w2s[j];
  }
  out[i] = acc;
}

// ---------------------------------------------------------------------------
extern "C" void kernel_launch(void* const* d_in, const int* in_sizes, int n_in,
                              void* d_out, int out_size, void* d_ws, size_t ws_size,
                              hipStream_t stream) {
  const float* x   = (const float*)d_in[0];
  const int*   ei  = (const int*)d_in[1];   // [2, E]
  const float* W1  = (const float*)d_in[2];
  const float* b1  = (const float*)d_in[3];
  const float* W2  = (const float*)d_in[4];
  const float* b2  = (const float*)d_in[5];
  const float* W3  = (const float*)d_in[6];
  const float* b3  = (const float*)d_in[7];
  const float* Wf1 = (const float*)d_in[8];
  const float* bf1 = (const float*)d_in[9];
  const float* Wf2 = (const float*)d_in[10];
  const float* bf2 = (const float*)d_in[11];

  const int IN_C = 256, H1 = 300, H2 = 100, H3 = 32;
  const int N = in_sizes[0] / IN_C;
  const int E = in_sizes[1] / 2;
  const int* row = ei;        // edge_index[0] = destination (segment ids)
  const int* col = ei + E;    // edge_index[1] = source

  // workspace layout: two N*H1 ping-pong buffers + dis[N]
  float* bufA = (float*)d_ws;
  float* bufB = bufA + (size_t)N * H1;
  float* dis  = bufB + (size_t)N * H1;

  const int TB = 256;
  const int GS = 2048;   // grid-stride block count for elementwise kernels
  const int GSC = 4096;  // for the big scatter

  // ---- degree -> dis = rsqrt(deg), self-loop counts as 1 ------------------
  fill_f32<<<(N + TB - 1) / TB, TB, 0, stream>>>(dis, N, 1.0f);
  deg_accum<<<1024, TB, 0, stream>>>(row, dis, E);
  make_dis<<<(N + TB - 1) / TB, TB, 0, stream>>>(dis, N);

  dim3 blk(128);
  auto grid_for = [&](int M, int F) { return dim3((M + TM - 1) / TM, (F + TN - 1) / TN); };

  // ---- layer 1: 256 -> 300 -----------------------------------------------
  gemm_wmma_bf16<<<grid_for(N, H1), blk, 0, stream>>>(x, W1, bufA, N, IN_C, H1);
  fill_f32<<<GS, TB, 0, stream>>>(bufB, (long)N * H1, 0.0f);
  scatter_edges<<<GSC, TB, 0, stream>>>(bufA, row, col, dis, bufB, E, H1);
  finalize_layer<<<GS, TB, 0, stream>>>(bufB, bufA, dis, b1, bufA, N, H1, 1);

  // ---- layer 2: 300 -> 100 -----------------------------------------------
  gemm_wmma_bf16<<<grid_for(N, H2), blk, 0, stream>>>(bufA, W2, bufB, N, H1, H2);
  fill_f32<<<GS, TB, 0, stream>>>(bufA, (long)N * H2, 0.0f);
  scatter_edges<<<GSC, TB, 0, stream>>>(bufB, row, col, dis, bufA, E, H2);
  finalize_layer<<<GS, TB, 0, stream>>>(bufA, bufB, dis, b2, bufB, N, H2, 1);

  // ---- layer 3: 100 -> 32 ------------------------------------------------
  gemm_wmma_bf16<<<grid_for(N, H3), blk, 0, stream>>>(bufB, W3, bufA, N, H2, H3);
  fill_f32<<<GS, TB, 0, stream>>>(bufB, (long)N * H3, 0.0f);
  scatter_edges<<<GSC, TB, 0, stream>>>(bufA, row, col, dis, bufB, E, H3);
  finalize_layer<<<GS, TB, 0, stream>>>(bufB, bufA, dis, b3, bufA, N, H3, 1);

  // ---- head: 32 -> 16 -> 1 (no relu) -------------------------------------
  head_mlp<<<(N + TB - 1) / TB, TB, 0, stream>>>(bufA, Wf1, bf1, Wf2, bf2,
                                                 (float*)d_out, N);
}
